// MultiHeadAttention_70016556860084
// MI455X (gfx1250) — compile-verified, matrix operand order checked by reading, not run on hardware
//
#include <hip/hip_runtime.h>

// ---------------------------------------------------------------------------
// MHA forward for MI455X (gfx1250, wave32, WMMA f32_16x16x32_f16)
// B=2, S=2048, D=1024, H=16, dk=64
// Pipeline:
//   1) fp32 -> f16 convert of q,k,v and Wq,Wk,Wv,Wo into workspace
//   2) WMMA GEMM (32x64 register-blocked): Qh = q@Wq^T+bq (head layout),
//      Kh likewise, Vt = (v@Wv^T+bv)^T per head
//   3) flash-attention (online softmax, mask, scale 1/sqrt(64)) -> ctx f16
//   4) WMMA GEMM: out = ctx@Wo^T + bo  (fp32 to d_out)
// ---------------------------------------------------------------------------

typedef __attribute__((ext_vector_type(16))) _Float16 v16h;
typedef __attribute__((ext_vector_type(8)))  _Float16 v8h;
typedef __attribute__((ext_vector_type(8)))  float    v8f;

#define Bsz 2
#define Ssz 2048
#define Dsz 1024
#define Hn  16
#define DK  64

__device__ __forceinline__ v8f wmma_f16(v16h a, v16h b, v8f c) {
  // v_wmma_f32_16x16x32_f16
  return __builtin_amdgcn_wmma_f32_16x16x32_f16(
      /*neg_a=*/false, a, /*neg_b=*/false, b,
      /*c_mod=*/(short)0, c, /*reuse_a=*/false, /*reuse_b=*/false);
}

// Load a 16-bit A/B fragment (16x32 row-per-lane) from a row-major matrix.
// rowptr points at the start of this lane's row; kb = K-block base.
// ISA layout: lanes 0-15 take k = kb+{0..7,16..23}; lanes 16-31 take +8.
__device__ __forceinline__ v16h ldfrag(const _Float16* rowptr, int kb, int laneHi) {
  const _Float16* p = rowptr + kb + laneHi * 8;
  v8h lo = *(const v8h*)(p);
  v8h hi = *(const v8h*)(p + 16);
  v16h f;
#pragma unroll
  for (int i = 0; i < 8; ++i) { f[i] = lo[i]; f[i + 8] = hi[i]; }
  return f;
}

__device__ __forceinline__ float rmax16(float v) {
#pragma unroll
  for (int m = 1; m < 16; m <<= 1) v = fmaxf(v, __shfl_xor(v, m, 32));
  return v;
}
__device__ __forceinline__ float rsum16(float v) {
#pragma unroll
  for (int m = 1; m < 16; m <<= 1) v += __shfl_xor(v, m, 32);
  return v;
}

// ---------------------------------------------------------------------------
__global__ void cvt_f32_f16(const float* __restrict__ in, _Float16* __restrict__ out, int n) {
  int i = blockIdx.x * blockDim.x + threadIdx.x;
  if (i < n) out[i] = (_Float16)in[i];
}

// ---------------------------------------------------------------------------
// C[M,N] = A[M,K] * Bt[N,K]^T + bias  (A,Bt f16 row-major, fp32 accumulate)
// M=4096, N=1024, K=1024.
// Register-blocked: one wave computes a 32x64 tile = 2 A-frags x 4 B-frags
// -> 8 WMMAs per K-step on 8 independent accumulator chains (hides XDL
// accumulation latency), 12 b128 loads per 8 WMMAs (vs 4 per WMMA naive).
// mode 0: f16 out in head layout   [b,h,s,dk]
// mode 1: f16 out transposed head  [b,h,dk,s]
// mode 2: fp32 out row-major       [M,N]
__global__ __launch_bounds__(256) void gemm_wmma_32x64(
    const _Float16* __restrict__ A, const _Float16* __restrict__ Bt,
    const float* __restrict__ bias, void* __restrict__ outp, int mode) {
  const int M = Bsz * Ssz, N = Dsz, K = Dsz;
  const int lane = threadIdx.x & 31;
  const int wave = threadIdx.x >> 5;
  const int tile = blockIdx.x * 8 + wave;
  const int nT64 = N / 64;                          // 16
  const int mT = tile / nT64;                       // 0..127 (32-row tiles)
  const int nT = tile % nT64;                       // 0..15  (64-col tiles)
  if (mT >= M / 32) return;
  const int ln = lane & 15, laneHi = lane >> 4;

  const _Float16* arow0 = A  + (size_t)(mT * 32 + ln)      * K;
  const _Float16* arow1 = A  + (size_t)(mT * 32 + 16 + ln) * K;
  const _Float16* brow0 = Bt + (size_t)(nT * 64 + ln)      * K;
  const _Float16* brow1 = Bt + (size_t)(nT * 64 + 16 + ln) * K;
  const _Float16* brow2 = Bt + (size_t)(nT * 64 + 32 + ln) * K;
  const _Float16* brow3 = Bt + (size_t)(nT * 64 + 48 + ln) * K;

  v8f c00 = {}, c01 = {}, c02 = {}, c03 = {};
  v8f c10 = {}, c11 = {}, c12 = {}, c13 = {};
#pragma unroll 4
  for (int kb = 0; kb < K; kb += 32) {
    const v16h a0 = ldfrag(arow0, kb, laneHi);
    const v16h a1 = ldfrag(arow1, kb, laneHi);
    const v16h b0 = ldfrag(brow0, kb, laneHi);
    const v16h b1 = ldfrag(brow1, kb, laneHi);
    const v16h b2 = ldfrag(brow2, kb, laneHi);
    const v16h b3 = ldfrag(brow3, kb, laneHi);
    c00 = wmma_f16(a0, b0, c00);  c01 = wmma_f16(a0, b1, c01);
    c02 = wmma_f16(a0, b2, c02);  c03 = wmma_f16(a0, b3, c03);
    c10 = wmma_f16(a1, b0, c10);  c11 = wmma_f16(a1, b1, c11);
    c12 = wmma_f16(a1, b2, c12);  c13 = wmma_f16(a1, b3, c13);
  }

  v8f acc[2][4] = {{c00, c01, c02, c03}, {c10, c11, c12, c13}};
#pragma unroll
  for (int mi = 0; mi < 2; ++mi) {
#pragma unroll
    for (int ni = 0; ni < 4; ++ni) {
      const int ng = nT * 64 + ni * 16 + ln;
      const float bvv = bias[ng];
      const v8f cc = acc[mi][ni];
#pragma unroll
      for (int i = 0; i < 8; ++i) {
        const int mg = mT * 32 + mi * 16 + i + laneHi * 8;   // C row per ISA layout
        const float val = cc[i] + bvv;
        if (mode == 2) {
          ((float*)outp)[(size_t)mg * N + ng] = val;
        } else {
          const int bb = mg >> 11, s = mg & (Ssz - 1);       // token -> (b,s)
          const int hh = ng >> 6,  inner = ng & (DK - 1);
          _Float16* o16 = (_Float16*)outp;
          if (mode == 0)
            o16[(((size_t)(bb * Hn + hh) * Ssz) + s) * DK + inner] = (_Float16)val;
          else
            o16[(((size_t)(bb * Hn + hh) * DK) + inner) * Ssz + s] = (_Float16)val;
        }
      }
    }
  }
}

// ---------------------------------------------------------------------------
// Flash attention: one wave per 16-query tile, 4 waves per block.
// Qh,Kh: [b,h,s,dk] f16    Vt: [b,h,dk,s] f16    ctx: [b,s,D] f16
__global__ __launch_bounds__(128) void flash_attn(
    const _Float16* __restrict__ Qh, const _Float16* __restrict__ Kh,
    const _Float16* __restrict__ Vt, const int* __restrict__ mask,
    _Float16* __restrict__ ctx) {
  const int lane = threadIdx.x & 31;
  const int wave = threadIdx.x >> 5;
  const int tile = blockIdx.x * 4 + wave;          // 0..4095
  const int qt = tile & (Ssz / 16 - 1);            // query tile in sequence
  const int bh = tile >> 7;                        // b*H + h
  const int b  = bh >> 4, h = bh & (Hn - 1);
  const int q0 = qt * 16;
  const int ln = lane & 15, laneHi = lane >> 4;

  const _Float16* Qb = Qh + (size_t)bh * Ssz * DK;
  const _Float16* Kb = Kh + (size_t)bh * Ssz * DK;
  const _Float16* Vb = Vt + (size_t)bh * DK * Ssz;
  const int*      mb = mask + (size_t)b * Ssz * Ssz;

  __shared__ _Float16 lds[4][16][32];              // per-wave P bounce buffer

  // Q fragments (held for the whole key sweep): 16 rows x dk=64
  const _Float16* qrow = Qb + (size_t)(q0 + ln) * DK;
  const v16h aq0 = ldfrag(qrow, 0,  laneHi);
  const v16h aq1 = ldfrag(qrow, 32, laneHi);

  v8f o0 = {}, o1 = {}, o2 = {}, o3 = {};
  float mr[8], lr[8];
#pragma unroll
  for (int i = 0; i < 8; ++i) { mr[i] = -3.0e38f; lr[i] = 0.0f; }

  for (int kb = 0; kb < Ssz; kb += 32) {
    // ---- scores: two 16x16 tiles (keys kb..kb+15, kb+16..kb+31), K=dk=64
    const _Float16* k0row = Kb + (size_t)(kb + ln) * DK;
    const _Float16* k1row = Kb + (size_t)(kb + 16 + ln) * DK;
    v8f s0 = {}, s1 = {};
    s0 = wmma_f16(aq0, ldfrag(k0row, 0,  laneHi), s0);
    s1 = wmma_f16(aq0, ldfrag(k1row, 0,  laneHi), s1);
    s0 = wmma_f16(aq1, ldfrag(k0row, 32, laneHi), s0);
    s1 = wmma_f16(aq1, ldfrag(k1row, 32, laneHi), s1);

    // ---- scale, mask, online softmax (row stats across 16-lane groups)
#pragma unroll
    for (int i = 0; i < 8; ++i) {
      const int qr = q0 + i + laneHi * 8;
      const int kc = kb + ln;
      float v0 = s0[i] * 0.125f;                   // 1/sqrt(dk)
      float v1 = s1[i] * 0.125f;
      if (mb[(size_t)qr * Ssz + kc] == 0)      v0 = -1.0e10f;
      if (mb[(size_t)qr * Ssz + kc + 16] == 0) v1 = -1.0e10f;
      const float cm    = rmax16(fmaxf(v0, v1));
      const float mnew  = fmaxf(mr[i], cm);
      const float alpha = __expf(mr[i] - mnew);
      const float p0 = __expf(v0 - mnew);
      const float p1 = __expf(v1 - mnew);
      lr[i] = lr[i] * alpha + rsum16(p0 + p1);
      mr[i] = mnew;
      o0[i] *= alpha; o1[i] *= alpha; o2[i] *= alpha; o3[i] *= alpha;
      // C-layout -> LDS (row-major 16x32 P tile, f16)
      lds[wave][i + laneHi * 8][ln]      = (_Float16)p0;
      lds[wave][i + laneHi * 8][ln + 16] = (_Float16)p1;
    }
    __syncthreads();
    const v16h pa = ldfrag(&lds[wave][ln][0], 0, laneHi);  // A-layout reload
    __syncthreads();

    // ---- O += P(16x32) * V(32x64); Vt rows are the B-matrix columns
    o0 = wmma_f16(pa, ldfrag(Vb + (size_t)(ln)      * Ssz, kb, laneHi), o0);
    o1 = wmma_f16(pa, ldfrag(Vb + (size_t)(16 + ln) * Ssz, kb, laneHi), o1);
    o2 = wmma_f16(pa, ldfrag(Vb + (size_t)(32 + ln) * Ssz, kb, laneHi), o2);
    o3 = wmma_f16(pa, ldfrag(Vb + (size_t)(48 + ln) * Ssz, kb, laneHi), o3);
  }

  // ---- normalize and write ctx (concat-head layout, f16)
#pragma unroll
  for (int i = 0; i < 8; ++i) {
    const float inv = 1.0f / lr[i];
    const int qr = q0 + i + laneHi * 8;
    _Float16* crow = ctx + ((size_t)(b * Ssz + qr)) * Dsz + h * DK;
    crow[ln]      = (_Float16)(o0[i] * inv);
    crow[16 + ln] = (_Float16)(o1[i] * inv);
    crow[32 + ln] = (_Float16)(o2[i] * inv);
    crow[48 + ln] = (_Float16)(o3[i] * inv);
  }
}

// ---------------------------------------------------------------------------
extern "C" void kernel_launch(void* const* d_in, const int* in_sizes, int n_in,
                              void* d_out, int out_size, void* d_ws, size_t ws_size,
                              hipStream_t stream) {
  (void)in_sizes; (void)n_in; (void)out_size; (void)ws_size;
  const float* q    = (const float*)d_in[0];
  const float* k    = (const float*)d_in[1];
  const float* v    = (const float*)d_in[2];
  const float* Wq   = (const float*)d_in[3];
  const float* bq   = (const float*)d_in[4];
  const float* Wk   = (const float*)d_in[5];
  const float* bk   = (const float*)d_in[6];
  const float* Wv   = (const float*)d_in[7];
  const float* bv   = (const float*)d_in[8];
  const float* Wo   = (const float*)d_in[9];
  const float* bo   = (const float*)d_in[10];
  const int*   mask = (const int*)d_in[11];

  // workspace layout (64 MB total)
  char* w = (char*)d_ws;
  const size_t MB = 1ull << 20;
  _Float16* q16  = (_Float16*)(w + 0 * MB);
  _Float16* k16  = (_Float16*)(w + 8 * MB);
  _Float16* v16f = (_Float16*)(w + 16 * MB);
  _Float16* wq16 = (_Float16*)(w + 24 * MB);
  _Float16* wk16 = (_Float16*)(w + 26 * MB);
  _Float16* wv16 = (_Float16*)(w + 28 * MB);
  _Float16* wo16 = (_Float16*)(w + 30 * MB);
  _Float16* QhP  = (_Float16*)(w + 32 * MB);
  _Float16* KhP  = (_Float16*)(w + 40 * MB);
  _Float16* VtP  = (_Float16*)(w + 48 * MB);
  _Float16* ctxP = (_Float16*)(w + 56 * MB);

  const int nAct = Bsz * Ssz * Dsz;   // 4M
  const int nW   = Dsz * Dsz;         // 1M
  cvt_f32_f16<<<(nAct + 255) / 256, 256, 0, stream>>>(q,  q16,  nAct);
  cvt_f32_f16<<<(nAct + 255) / 256, 256, 0, stream>>>(k,  k16,  nAct);
  cvt_f32_f16<<<(nAct + 255) / 256, 256, 0, stream>>>(v,  v16f, nAct);
  cvt_f32_f16<<<(nW + 255) / 256, 256, 0, stream>>>(Wq, wq16, nW);
  cvt_f32_f16<<<(nW + 255) / 256, 256, 0, stream>>>(Wk, wk16, nW);
  cvt_f32_f16<<<(nW + 255) / 256, 256, 0, stream>>>(Wv, wv16, nW);
  cvt_f32_f16<<<(nW + 255) / 256, 256, 0, stream>>>(Wo, wo16, nW);

  // (M/32)*(N/64) = 128*16 = 2048 wave-tiles, 8 waves per 256-thread block
  const int gemmBlocks = 2048 / 8;
  gemm_wmma_32x64<<<gemmBlocks, 256, 0, stream>>>(q16,  wq16, bq, (void*)QhP, 0);
  gemm_wmma_32x64<<<gemmBlocks, 256, 0, stream>>>(k16,  wk16, bk, (void*)KhP, 0);
  gemm_wmma_32x64<<<gemmBlocks, 256, 0, stream>>>(v16f, wv16, bv, (void*)VtP, 1);

  // B*H*(S/16) = 4096 query tiles, 4 waves per 128-thread block
  flash_attn<<<4096 / 4, 128, 0, stream>>>(QhP, KhP, VtP, mask, ctxP);

  gemm_wmma_32x64<<<gemmBlocks, 256, 0, stream>>>(ctxP, wo16, bo, d_out, 2);
}
